// ARTeacher_55147380081191
// MI455X (gfx1250) — matrix operand order repaired and checked by reading, not compiled, not run on hardware
//
#include <hip/hip_runtime.h>
#include <math.h>

typedef __attribute__((ext_vector_type(16))) _Float16 v16h;
typedef __attribute__((ext_vector_type(8)))  _Float16 v8h;
typedef __attribute__((ext_vector_type(8)))  float    v8f;
typedef int v4i_ __attribute__((vector_size(16)));

#if defined(__has_builtin)
#if __has_builtin(__builtin_amdgcn_global_load_async_to_lds_b128) && \
    __has_builtin(__builtin_amdgcn_s_wait_asynccnt)
#define USE_ASYNC_COPY 1
#else
#define USE_ASYNC_COPY 0
#endif
#else
#define USE_ASYNC_COPY 0
#endif

namespace {
constexpr int VSZ = 8000, D = 512, L = 8, NH = 8, BB = 16, PROMPT = 64, TNEW = 16;
constexpr int T  = PROMPT + TNEW;   // 80
constexpr int HD = D / NH;          // 64
constexpr int PAD_ID = 0, EOS_ID = 3;
constexpr int M_ROWS = BB * T;      // 1280
constexpr int BM = 128, BN = 64;    // GEMM block tile (4 waves, each 32x64)
}

#if USE_ASYNC_COPY
__device__ __forceinline__ void async_cp16(const _Float16* g, _Float16* l) {
  // GLOBAL_LOAD_ASYNC_TO_LDS_B128: memory -> LDS, no VGPR round-trip, ASYNCcnt.
  __builtin_amdgcn_global_load_async_to_lds_b128(
      (__attribute__((address_space(1))) v4i_*)g,
      (__attribute__((address_space(3))) v4i_*)l, 0, 0);
}
#endif

// ---------------- f32 -> f16 transpose:  W[K][N] -> Wt[N][K] ----------------
__global__ __launch_bounds__(256)
void k_transpose_f16(const float* __restrict__ W, _Float16* __restrict__ Wt, int K, int N) {
  __shared__ float tile[32][33];
  const int n0 = blockIdx.x * 32, k0 = blockIdx.y * 32;
  const int tx = threadIdx.x & 31, ty = threadIdx.x >> 5;
  for (int i = ty; i < 32; i += 8) {
    int k = k0 + i, n = n0 + tx;
    tile[i][tx] = (k < K && n < N) ? W[(size_t)k * N + n] : 0.f;
  }
  __syncthreads();
  for (int i = ty; i < 32; i += 8) {
    int n = n0 + i, k = k0 + tx;
    if (n < N && k < K) Wt[(size_t)n * K + k] = (_Float16)tile[tx][i];
  }
}

// ---------------- decode-state init ----------------
__global__ __launch_bounds__(256)
void k_init(const int* __restrict__ ids_in, const int* __restrict__ mask_in,
            int* ids, int* msk, int* lens, int* fin, float* csum, float* ccnt) {
  const int tid = threadIdx.x;
  for (int i = tid; i < BB * T; i += 256) { ids[i] = ids_in[i]; msk[i] = mask_in[i]; }
  if (tid < BB) {
    int s = 0;
    for (int t = 0; t < T; ++t) s += (mask_in[tid * T + t] != 0);
    lens[tid] = s; fin[tid] = 0; csum[tid] = 0.f; ccnt[tid] = 0.f;
  }
}

// ---------------- token + position embedding ----------------
__global__ __launch_bounds__(256)
void k_embed(const int* __restrict__ ids, const float* __restrict__ wte,
             const float* __restrict__ wpe, float* __restrict__ x) {
  const int row = blockIdx.x, tid = threadIdx.x;
  const int t = row % T;
  const int id = ids[row];
  x[(size_t)row * D + tid]       = wte[(size_t)id * D + tid]       + wpe[(size_t)t * D + tid];
  x[(size_t)row * D + tid + 256] = wte[(size_t)id * D + tid + 256] + wpe[(size_t)t * D + tid + 256];
}

// ---------------- LayerNorm (f32 in, f16 out for GEMM A) ----------------
__global__ __launch_bounds__(256)
void k_layernorm(const float* __restrict__ x, const float* __restrict__ w,
                 const float* __restrict__ bset, _Float16* __restrict__ out) {
  __shared__ float s1[256], s2[256];
  const int row = blockIdx.x, tid = threadIdx.x;
  const float a = x[(size_t)row * D + tid];
  const float c = x[(size_t)row * D + tid + 256];
  s1[tid] = a + c; s2[tid] = a * a + c * c;
  __syncthreads();
  for (int s = 128; s > 0; s >>= 1) {
    if (tid < s) { s1[tid] += s1[tid + s]; s2[tid] += s2[tid + s]; }
    __syncthreads();
  }
  const float mean = s1[0] * (1.f / D);
  const float var  = s2[0] * (1.f / D) - mean * mean;
  const float inv  = rsqrtf(var + 1e-5f);
  out[(size_t)row * D + tid]       = (_Float16)((a - mean) * inv * w[tid] + bset[tid]);
  out[(size_t)row * D + tid + 256] = (_Float16)((c - mean) * inv * w[tid + 256] + bset[tid + 256]);
}

// ---------------- WMMA GEMM:  out[M,N] = A[M,K] * Bt[N,K]^T (+bias/epilogue) ----
// mode 0: outF = acc + bias
// mode 1: outF += acc + bias     (residual, in place)
// mode 2: outH = f16(gelu(acc + bias))
// 128 threads = 4 waves; block tile 128x64; each wave computes 32x64
// (2 A-frags x 4 B-frags = 8 WMMA per K-step). Double-buffered LDS fed by
// async global->LDS copies pipelined one K-tile ahead of the WMMA consumer.
__global__ __launch_bounds__(128)
void k_gemm(const _Float16* __restrict__ A, int lda,
            const _Float16* __restrict__ Bt, int ldb,
            const float* __restrict__ bias,
            float* outF, _Float16* outH, int M, int N, int K, int mode) {
  __shared__ _Float16 lA[2][BM * 40];   // 128 rows x 32 K, stride 40 (bank stagger)
  __shared__ _Float16 lB[2][BN * 40];   //  64 rows x 32 K
  const int tid = threadIdx.x, lane = tid & 31, w = tid >> 5;
  const int tM = blockIdx.y * BM, tN = blockIdx.x * BN;
  v8f acc[2][4] = {};
  const int offA = (lane < 16) ? 0 : 8;    // A frag: K 0..7/16..23 vs 8..15/24..31
  const int offB = (lane < 16) ? 0 : 8;    // B frag: K 0..7/16..23 vs 8..15/24..31
  const int ar0 = (w << 5) + (lane & 15);  // wave rows: [w*32, w*32+32)
  const int ar1 = ar0 + 16;

  // cooperative-copy chunk coordinates (16B chunks): A 4/thread, B 2/thread
  const int rr0 = tid >> 2, c0 = (tid & 3) * 8;    // rows 0..31 (+32k)
  size_t aOff[4];
  int lOffA[4];
#pragma unroll
  for (int q = 0; q < 4; ++q) {
    int gr = tM + rr0 + 32 * q; if (gr >= M) gr = M - 1;   // clamp (LM head M=16)
    aOff[q] = (size_t)gr * lda + c0;
    lOffA[q] = (rr0 + 32 * q) * 40 + c0;
  }
  const size_t bOff0 = (size_t)(tN + rr0) * ldb + c0;
  const size_t bOff1 = (size_t)(tN + rr0 + 32) * ldb + c0;
  const int lOffB0 = rr0 * 40 + c0, lOffB1 = (rr0 + 32) * 40 + c0;

  const int nk = K >> 5;
#if USE_ASYNC_COPY
#pragma unroll
  for (int q = 0; q < 4; ++q) async_cp16(A + aOff[q], &lA[0][lOffA[q]]);
  async_cp16(Bt + bOff0, &lB[0][lOffB0]);
  async_cp16(Bt + bOff1, &lB[0][lOffB1]);
#endif
  for (int i = 0; i < nk; ++i) {
    const int k0 = i << 5;
    const int buf = i & 1;
#if USE_ASYNC_COPY
    if (i + 1 < nk) {          // pipeline tile i+1 into the other buffer
      const int kn = k0 + 32, bn = buf ^ 1;
#pragma unroll
      for (int q = 0; q < 4; ++q) async_cp16(A + aOff[q] + kn, &lA[bn][lOffA[q]]);
      async_cp16(Bt + bOff0 + kn, &lB[bn][lOffB0]);
      async_cp16(Bt + bOff1 + kn, &lB[bn][lOffB1]);
      __builtin_amdgcn_s_wait_asynccnt(6);   // async loads retire in order:
    } else {                                 // <=6 outstanding => tile i landed
      __builtin_amdgcn_s_wait_asynccnt(0);
    }
#else
    {
      // issue all global loads before any LDS store (avoid per-chunk stall)
      v8h va[4], vb0, vb1;
#pragma unroll
      for (int q = 0; q < 4; ++q) va[q] = *(const v8h*)(A + aOff[q] + k0);
      vb0 = *(const v8h*)(Bt + bOff0 + k0);
      vb1 = *(const v8h*)(Bt + bOff1 + k0);
      if (k0 + 32 < K) {
        __builtin_prefetch(A + aOff[0] + k0 + 32, 0, 1);
        __builtin_prefetch(Bt + bOff0 + k0 + 32, 0, 1);
      }
#pragma unroll
      for (int q = 0; q < 4; ++q) *(v8h*)&lA[buf][lOffA[q]] = va[q];
      *(v8h*)&lB[buf][lOffB0] = vb0;
      *(v8h*)&lB[buf][lOffB1] = vb1;
    }
#endif
    __syncthreads();

    // two A fragments (16x32 f16 each, documented CDNA5 layout)
    const v8h a0lo = *(const v8h*)&lA[buf][ar0 * 40 + offA];
    const v8h a0hi = *(const v8h*)&lA[buf][ar0 * 40 + offA + 16];
    const v16h af0 = __builtin_shufflevector(a0lo, a0hi, 0,1,2,3,4,5,6,7,8,9,10,11,12,13,14,15);
    const v8h a1lo = *(const v8h*)&lA[buf][ar1 * 40 + offA];
    const v8h a1hi = *(const v8h*)&lA[buf][ar1 * 40 + offA + 16];
    const v16h af1 = __builtin_shufflevector(a1lo, a1hi, 0,1,2,3,4,5,6,7,8,9,10,11,12,13,14,15);
#pragma unroll
    for (int j = 0; j < 4; ++j) {
      const int br = (j << 4) + (lane & 15);
      const v8h blo = *(const v8h*)&lB[buf][br * 40 + offB];
      const v8h bhi = *(const v8h*)&lB[buf][br * 40 + offB + 16];
      const v16h bf = __builtin_shufflevector(blo, bhi, 0,1,2,3,4,5,6,7,8,9,10,11,12,13,14,15);
      acc[0][j] = __builtin_amdgcn_wmma_f32_16x16x32_f16(false, af0, false, bf, (short)0,
                                                         acc[0][j], false, false);
      acc[1][j] = __builtin_amdgcn_wmma_f32_16x16x32_f16(false, af1, false, bf, (short)0,
                                                         acc[1][j], false, false);
    }
    __syncthreads();
  }

  const int cbase = tN + (lane & 15);
#pragma unroll
  for (int a = 0; a < 2; ++a) {
    const int rbase = tM + (w << 5) + (a << 4) + ((lane < 16) ? 0 : 8);
#pragma unroll
    for (int j = 0; j < 4; ++j) {
      const int gc = cbase + (j << 4);
      const float bv = bias ? bias[gc] : 0.f;
#pragma unroll
      for (int rr = 0; rr < 8; ++rr) {
        const int gr = rbase + rr;
        if (gr >= M) continue;
        float v = acc[a][j][rr] + bv;
        if (mode == 1) v += outF[(size_t)gr * N + gc];
        if (mode == 2) {
          const float t = 0.7978845608028654f * (v + 0.044715f * v * v * v);
          outH[(size_t)gr * N + gc] = (_Float16)(0.5f * v * (1.f + tanhf(t)));
        } else {
          outF[(size_t)gr * N + gc] = v;
        }
      }
    }
  }
}

// ---------------- attention (one wave per (b,h,q) row; ~1% of FLOPs) --------
__global__ __launch_bounds__(128)
void k_attn(const float* __restrict__ qkv, const int* __restrict__ msk,
            _Float16* __restrict__ attnH) {
  __shared__ float sc[4][T];
  const int lane = threadIdx.x & 31, w = threadIdx.x >> 5;
  const int r = blockIdx.x * 4 + w;           // [0, B*H*T)
  const int b = r / (NH * T);
  const int h = (r / T) % NH;
  const int q = r % T;
  const int d0 = lane * 2;
  const size_t qrow = (size_t)(b * T + q) * (3 * D);
  const float q0 = qkv[qrow + h * HD + d0];
  const float q1 = qkv[qrow + h * HD + d0 + 1];
  for (int t = 0; t < T; ++t) {
    const float* kp = qkv + (size_t)(b * T + t) * (3 * D) + D + h * HD + d0;
    float partial = q0 * kp[0] + q1 * kp[1];
    for (int off = 16; off >= 1; off >>= 1) partial += __shfl_xor(partial, off, 32);
    float s = partial * 0.125f;                        // 1/sqrt(64)
    if (!(t <= q && msk[b * T + t] > 0)) s = -1e9f;
    if (lane == 0) sc[w][t] = s;
  }
  __syncthreads();
  float m = -3.0e38f;
  for (int t = lane; t < T; t += 32) m = fmaxf(m, sc[w][t]);
  for (int off = 16; off >= 1; off >>= 1) m = fmaxf(m, __shfl_xor(m, off, 32));
  float sum = 0.f;
  for (int t = lane; t < T; t += 32) {
    const float e = __expf(sc[w][t] - m);
    sc[w][t] = e;
    sum += e;
  }
  for (int off = 16; off >= 1; off >>= 1) sum += __shfl_xor(sum, off, 32);
  __syncthreads();
  float a0 = 0.f, a1 = 0.f;
  for (int t = 0; t < T; ++t) {
    const float pb = sc[w][t];
    const float* vp = qkv + (size_t)(b * T + t) * (3 * D) + 2 * D + h * HD + d0;
    a0 += pb * vp[0];
    a1 += pb * vp[1];
  }
  const float inv = 1.f / sum;
  attnH[(size_t)(b * T + q) * D + h * HD + d0]     = (_Float16)(a0 * inv);
  attnH[(size_t)(b * T + q) * D + h * HD + d0 + 1] = (_Float16)(a1 * inv);
}

// ---------------- gather last-token hidden rows for LM head ----------------
__global__ __launch_bounds__(256)
void k_gather(const _Float16* __restrict__ hf, const int* __restrict__ lens,
              _Float16* __restrict__ afin) {
  const int b = blockIdx.x, tid = threadIdx.x;
  int last = lens[b] - 1; if (last < 0) last = 0; if (last > T - 1) last = T - 1;
  const size_t src = (size_t)(b * T + last) * D;
  afin[(size_t)b * D + tid]       = hf[src + tid];
  afin[(size_t)b * D + tid + 256] = hf[src + tid + 256];
}

// ---------------- greedy sample + confidence + state update ----------------
__global__ __launch_bounds__(256)
void k_sample(const float* __restrict__ logits, float* __restrict__ outTok,
              int* ids, int* msk, int* lens, int* fin, float* csum, float* ccnt, int step) {
  __shared__ float sv[256]; __shared__ int si[256]; __shared__ float ss[256];
  const int b = blockIdx.x, tid = threadIdx.x;
  float best = -3.0e38f; int bi = VSZ;
  for (int j = tid; j < VSZ; j += 256) {
    const float v = logits[(size_t)b * VSZ + j];
    if (v > best || (v == best && j < bi)) { best = v; bi = j; }
  }
  sv[tid] = best; si[tid] = bi; __syncthreads();
  for (int s = 128; s > 0; s >>= 1) {
    if (tid < s) {
      const float v = sv[tid + s]; const int i2 = si[tid + s];
      if (v > sv[tid] || (v == sv[tid] && i2 < si[tid])) { sv[tid] = v; si[tid] = i2; }
    }
    __syncthreads();
  }
  const float mx = sv[0]; const int top = si[0];
  float le = 0.f;
  for (int j = tid; j < VSZ; j += 256) le += __expf(logits[(size_t)b * VSZ + j] - mx);
  ss[tid] = le; __syncthreads();
  for (int s = 128; s > 0; s >>= 1) { if (tid < s) ss[tid] += ss[tid + s]; __syncthreads(); }
  if (tid == 0) {
    const float top_p = 1.f / ss[0];           // exp(mx-mx)/sumexp
    const int ln = lens[b];
    const bool active = (fin[b] == 0) && (ln < T);
    if (active) { csum[b] += top_p; ccnt[b] += 1.f; }
    const bool is_end = (top == EOS_ID) || (top == PAD_ID);
    const bool wr = active && !is_end;
    int pos = ln; if (pos < 0) pos = 0; if (pos > T - 1) pos = T - 1;
    if (wr) { ids[b * T + pos] = top; msk[b * T + pos] = 1; lens[b] = ln + 1; }
    if (active && is_end) fin[b] = 1;
    outTok[b * TNEW + step] = (float)(wr ? top : PAD_ID);
  }
}

__global__ void k_finalize(const float* csum, const float* ccnt, float* out) {
  const int b = threadIdx.x;
  if (b < BB) out[b] = csum[b] / fmaxf(ccnt[b], 1.f);
}

// ======================= host-side launch sequence =========================
extern "C" void kernel_launch(void* const* d_in, const int* in_sizes, int n_in,
                              void* d_out, int out_size, void* d_ws, size_t ws_size,
                              hipStream_t stream) {
  (void)in_sizes; (void)n_in; (void)out_size; (void)ws_size;
  const float* wte    = (const float*)d_in[0];
  const float* wpe    = (const float*)d_in[1];
  const float* ln1_w  = (const float*)d_in[2];
  const float* ln1_b  = (const float*)d_in[3];
  const float* qkv_w  = (const float*)d_in[4];
  const float* qkv_b  = (const float*)d_in[5];
  const float* ao_w   = (const float*)d_in[6];
  const float* ao_b   = (const float*)d_in[7];
  const float* ln2_w  = (const float*)d_in[8];
  const float* ln2_b  = (const float*)d_in[9];
  const float* fc_w   = (const float*)d_in[10];
  const float* fc_b   = (const float*)d_in[11];
  const float* pr_w   = (const float*)d_in[12];
  const float* pr_b   = (const float*)d_in[13];
  const float* lnf_w  = (const float*)d_in[14];
  const float* lnf_b  = (const float*)d_in[15];
  const float* lm_w   = (const float*)d_in[16];
  const int* input_ids = (const int*)d_in[17];
  const int* attn_mask = (const int*)d_in[18];

  char* p = (char*)d_ws;
  auto take = [&](size_t bytes) -> char* {
    char* r = p; p += (bytes + 255) & ~(size_t)255; return r;
  };
  _Float16* wQKV  = (_Float16*)take((size_t)L * 3 * D * D * 2);   // [l][3D][D]
  _Float16* wAO   = (_Float16*)take((size_t)L * D * D * 2);       // [l][D][D]
  _Float16* wFC   = (_Float16*)take((size_t)L * 4 * D * D * 2);   // [l][4D][D]
  _Float16* wPR   = (_Float16*)take((size_t)L * D * 4 * D * 2);   // [l][D][4D]
  _Float16* wLM   = (_Float16*)take((size_t)VSZ * D * 2);         // [V][D]
  float*    x     = (float*)take((size_t)M_ROWS * D * 4);
  _Float16* hbuf  = (_Float16*)take((size_t)M_ROWS * D * 2);
  float*    qkvB  = (float*)take((size_t)M_ROWS * 3 * D * 4);
  _Float16* attnH = (_Float16*)take((size_t)M_ROWS * D * 2);
  _Float16* ubuf  = (_Float16*)take((size_t)M_ROWS * 4 * D * 2);
  _Float16* afin  = (_Float16*)take((size_t)BB * D * 2);
  float*    logits= (float*)take((size_t)BB * VSZ * 4);
  int*      idsWs = (int*)take((size_t)BB * T * 4);
  int*      mskWs = (int*)take((size_t)BB * T * 4);
  int*      lensWs= (int*)take(64);
  int*      finWs = (int*)take(64);
  float*    csumWs= (float*)take(64);
  float*    ccntWs= (float*)take(64);
  float*    out   = (float*)d_out;

  // --- weight conversion + transpose to [N][K] f16 (58 MB: resident in 192MB L2) ---
  for (int l = 0; l < L; ++l) {
    k_transpose_f16<<<dim3(3 * D / 32, D / 32), 256, 0, stream>>>(
        qkv_w + (size_t)l * D * 3 * D, wQKV + (size_t)l * 3 * D * D, D, 3 * D);
    k_transpose_f16<<<dim3(D / 32, D / 32), 256, 0, stream>>>(
        ao_w + (size_t)l * D * D, wAO + (size_t)l * D * D, D, D);
    k_transpose_f16<<<dim3(4 * D / 32, D / 32), 256, 0, stream>>>(
        fc_w + (size_t)l * D * 4 * D, wFC + (size_t)l * 4 * D * D, D, 4 * D);
    k_transpose_f16<<<dim3(D / 32, 4 * D / 32), 256, 0, stream>>>(
        pr_w + (size_t)l * 4 * D * D, wPR + (size_t)l * D * 4 * D, 4 * D, D);
  }
  k_transpose_f16<<<dim3(VSZ / 32, D / 32), 256, 0, stream>>>(lm_w, wLM, D, VSZ);

  k_init<<<1, 256, 0, stream>>>(input_ids, attn_mask, idsWs, mskWs, lensWs, finWs,
                                csumWs, ccntWs);

  const dim3 blk(128);
  const int gM = M_ROWS / BM;     // 10
  for (int step = 0; step < TNEW; ++step) {
    k_embed<<<M_ROWS, 256, 0, stream>>>(idsWs, wte, wpe, x);
    for (int l = 0; l < L; ++l) {
      k_layernorm<<<M_ROWS, 256, 0, stream>>>(x, ln1_w + l * D, ln1_b + l * D, hbuf);
      k_gemm<<<dim3(3 * D / BN, gM), blk, 0, stream>>>(
          hbuf, D, wQKV + (size_t)l * 3 * D * D, D, qkv_b + l * 3 * D,
          qkvB, nullptr, M_ROWS, 3 * D, D, 0);
      k_attn<<<(BB * NH * T) / 4, 128, 0, stream>>>(qkvB, mskWs, attnH);
      k_gemm<<<dim3(D / BN, gM), blk, 0, stream>>>(
          attnH, D, wAO + (size_t)l * D * D, D, ao_b + l * D,
          x, nullptr, M_ROWS, D, D, 1);
      k_layernorm<<<M_ROWS, 256, 0, stream>>>(x, ln2_w + l * D, ln2_b + l * D, hbuf);
      k_gemm<<<dim3(4 * D / BN, gM), blk, 0, stream>>>(
          hbuf, D, wFC + (size_t)l * 4 * D * D, D, fc_b + l * 4 * D,
          nullptr, ubuf, M_ROWS, 4 * D, D, 2);
      k_gemm<<<dim3(D / BN, gM), blk, 0, stream>>>(
          ubuf, 4 * D, wPR + (size_t)l * D * 4 * D, 4 * D, pr_b + l * D,
          x, nullptr, M_ROWS, D, 4 * D, 1);
    }
    k_layernorm<<<M_ROWS, 256, 0, stream>>>(x, lnf_w, lnf_b, hbuf);
    k_gather<<<BB, 256, 0, stream>>>(hbuf, lensWs, afin);
    k_gemm<<<dim3(VSZ / BN, 1), blk, 0, stream>>>(
        afin, D, wLM, D, nullptr, logits, nullptr, BB, VSZ, D, 0);
    k_sample<<<BB, 256, 0, stream>>>(logits, out, idsWs, mskWs, lensWs, finWs,
                                     csumWs, ccntWs, step);
  }
  k_finalize<<<1, 64, 0, stream>>>(csumWs, ccntWs, out + BB * TNEW);
}
